// MultiHeadAttention_39926015983741
// MI455X (gfx1250) — matrix operand order repaired and checked by reading, not compiled, not run on hardware
//
#include <hip/hip_runtime.h>

// ---------------------------------------------------------------------------
// MHA for gfx1250 (MI455X): f16 WMMA (v_wmma_f32_16x16x32_f16), f32 accumulate.
// CDNA5 paths: v_wmma_*, global_load_async_to_lds_b128 (+ s_wait_asynccnt),
// global_prefetch_b8, wide ds_load_b128 fragment reads via frag-order staging.
// Workspace: Qh,Kh,Vh [B,H,L,D] f16 + Z [B,L,C] f16 = 4 * 8 MB = 32 MB.
// ---------------------------------------------------------------------------

typedef __attribute__((ext_vector_type(16))) _Float16 v16h;
typedef __attribute__((ext_vector_type(8)))  _Float16 v8h;
typedef __attribute__((ext_vector_type(4)))  _Float16 v4h;
typedef __attribute__((ext_vector_type(8)))  float    v8f;

constexpr int Bsz  = 2;
constexpr int Lseq = 2048;
constexpr int Cdim = 1024;
constexpr int Hn   = 16;
constexpr int Dh   = 64;
constexpr int Mtot = Bsz * Lseq;   // 4096

__device__ __forceinline__ v16h concat8(v8h lo, v8h hi) {
  return __builtin_shufflevector(lo, hi, 0, 1, 2, 3, 4, 5, 6, 7,
                                 8, 9, 10, 11, 12, 13, 14, 15);
}

// Async 16B/lane global->LDS copy (ASYNCcnt). INST_OFFSET applies to BOTH the
// LDS and the global address, so callers arrange equal strides on both sides.
template <int OFF>
__device__ __forceinline__ void async_copy_b128(unsigned lds_addr, const void* gaddr) {
  asm volatile("global_load_async_to_lds_b128 %0, %1, off offset:%2"
               :: "v"(lds_addr), "v"((unsigned long long)gaddr), "i"(OFF) : "memory");
}
__device__ __forceinline__ void wait_asynccnt0() {
  asm volatile("s_wait_asynccnt 0x0" ::: "memory");
}

union DW4 { uint4 u4; _Float16 h[8]; };

// Fragment maps (16-bit WMMA):
//  A 16x32: lane holds row (lane&15); elems = one contiguous 16-run at
//           K = half*16 within each 32-K chunk (same k order as B).
//  B 32x16: lane (n&15)+16*(kk>>4) holds K=kk, elem j = kk&15 (contiguous 32B).
//  C/D:     elem v -> row v + 8*half, col = lane&15.

// ---------------------------------------------------------------------------
// Kernel 1: Y[b,h,l,d] = (X @ W + bias) scattered head-major, f16 out.
// 128 threads (4 waves). Tile 64(M) x 64(N), K-steps of 64.
// ---------------------------------------------------------------------------
__global__ void __launch_bounds__(128)
qkv_proj_kernel(const float* __restrict__ X, const float* __restrict__ W,
                const float* __restrict__ bias, _Float16* __restrict__ out)
{
  __shared__ _Float16 lX[64][64];          // A tile, row-major
  __shared__ _Float16 lWf[4][2][32][16];   // B tile in fragment order
  const int tid = threadIdx.x;
  const int wv = tid >> 5, ln = tid & 31, lh = ln >> 4, lm = ln & 15;
  const int m0 = blockIdx.x * 64;
  const int n0 = blockIdx.y * 64;

  // per-thread staging bases (row = tid>>4, 4-float chunk = tid&15)
  const int sr = tid >> 4, sc4 = (tid & 15) << 2;
  _Float16* sX = &lX[sr][sc4];

  v8f acc[4] = {};

  for (int k0 = 0; k0 < Cdim; k0 += 64) {
    // stage X tile 64x64 (f32 -> f16): base + constant offsets
    const float4* pX = (const float4*)&X[(size_t)(m0 + sr) * Cdim + k0 + sc4];
    #pragma unroll
    for (int it = 0; it < 8; ++it) {
      float4 f = pX[it * (8 * Cdim / 4)];            // +it*8 rows
      v4h p = { (_Float16)f.x, (_Float16)f.y, (_Float16)f.z, (_Float16)f.w };
      *(v4h*)(sX + it * 8 * 64) = p;
    }
    // stage W tile 64x64 (f32 -> f16) scattered into B-fragment order
    const float4* pW = (const float4*)&W[(size_t)(k0 + sr) * Cdim + n0 + sc4];
    #pragma unroll
    for (int it = 0; it < 8; ++it) {
      float4 f = pW[it * (8 * Cdim / 4)];
      int r = sr + it * 8;                           // k within tile
      int c = r >> 5, kk = r & 31;
      int laneBase = 16 * (kk >> 4), j = kk & 15;
      float fv[4] = { f.x, f.y, f.z, f.w };
      #pragma unroll
      for (int u = 0; u < 4; ++u) {
        int n = sc4 + u;
        lWf[n >> 4][c][laneBase + (n & 15)][j] = (_Float16)fv[u];
      }
    }
    if (k0 + 64 < Cdim) {  // global_prefetch_b8 hints
      __builtin_prefetch(&X[(size_t)(m0 + (tid >> 1)) * Cdim + k0 + 64], 0, 3);
      __builtin_prefetch(&W[(size_t)(k0 + 64 + (tid >> 1)) * Cdim + n0], 0, 3);
    }
    __syncthreads();

    #pragma unroll
    for (int c = 0; c < 2; ++c) {
      v8h alo = *(const v8h*)&lX[wv * 16 + lm][c * 32 + lh * 16];
      v8h ahi = *(const v8h*)&lX[wv * 16 + lm][c * 32 + lh * 16 + 8];
      v16h a = concat8(alo, ahi);
      v16h bfr[4];
      #pragma unroll
      for (int t = 0; t < 4; ++t)            // 8 x ds_load_b128, one wait
        bfr[t] = *(const v16h*)&lWf[t][c][ln][0];
      #pragma unroll
      for (int t = 0; t < 4; ++t)
        acc[t] = __builtin_amdgcn_wmma_f32_16x16x32_f16(
            false, a, false, bfr[t], (short)0, acc[t], false, false);
    }
    __syncthreads();
  }

  // epilogue: bias + head-major scatter
  #pragma unroll
  for (int t = 0; t < 4; ++t) {
    #pragma unroll
    for (int v = 0; v < 8; ++v) {
      int m = m0 + wv * 16 + v + 8 * lh;
      int n = n0 + t * 16 + lm;
      float val = acc[t][v] + bias[n];
      int b = m >> 11, l = m & (Lseq - 1);
      int h = n >> 6,  d = n & (Dh - 1);
      out[(((size_t)b * Hn + h) * Lseq + l) * Dh + d] = (_Float16)val;
    }
  }
}

// ---------------------------------------------------------------------------
// Kernel 2: flash attention per (b,h). 4 waves x 16 query rows = 64 q/block.
// Key blocks of 64: scores = Q K^T (8 WMMA), online softmax, O += P V (8 WMMA).
// K block staged via global_load_async_to_lds_b128 (ASYNCcnt).
// ---------------------------------------------------------------------------
__global__ void __launch_bounds__(128)
attn_kernel(const _Float16* __restrict__ Qh, const _Float16* __restrict__ Kh,
            const _Float16* __restrict__ Vh, const int* __restrict__ mask,
            _Float16* __restrict__ Z)
{
  __shared__ _Float16 lK[64][64];          // K block, row-major (rows = keys)
  __shared__ _Float16 lVf[4][2][32][16];   // V block in B-fragment order
  __shared__ _Float16 lP[4][16][64];       // per-wave P staging

  const int tid = threadIdx.x;
  const int wv = tid >> 5, ln = tid & 31, lh = ln >> 4, lm = ln & 15;
  const int bh = blockIdx.x;
  const int b = bh / Hn, h = bh % Hn;
  const int q0 = blockIdx.y * 64 + wv * 16;
  const size_t headBase = (size_t)bh * Lseq * Dh;
  // per-thread K-copy base: LDS and global advance by 2048 B per step
  const unsigned ldsKt = (unsigned)(size_t)&lK[0][0] + (unsigned)tid * 16u;
  const char* gKt = (const char*)(Kh + headBase) + tid * 16;
  // per-thread V-staging base (row = tid>>3, 8-f16 chunk = tid&7)
  const int vk0 = tid >> 3, vd0 = (tid & 7) << 3;
  const uint4* pV0 = (const uint4*)(Vh + headBase) + tid;
  // per-thread mask base: row (q0 + 8*lh + v), col (key0 + nt*16 + lm)
  const int* pm = mask + ((size_t)b * Lseq + q0 + 8 * lh) * Lseq + lm;

  // Q fragments pinned in registers: 16 rows x 64 d = 2 A-frags (global b128s)
  v16h aQ[2];
  #pragma unroll
  for (int c = 0; c < 2; ++c) {
    const _Float16* qrow = Qh + headBase + (size_t)(q0 + lm) * Dh + c * 32;
    aQ[c] = concat8(*(const v8h*)&qrow[lh * 16], *(const v8h*)&qrow[lh * 16 + 8]);
  }

  v8f O[4] = {};
  float mrow[8], lrow[8];
  #pragma unroll
  for (int v = 0; v < 8; ++v) { mrow[v] = -3.0e38f; lrow[v] = 0.0f; }

  for (int key0 = 0; key0 < Lseq; key0 += 64) {
    // ---- K block: async copy global -> LDS (8 KB; 1 base + imm offsets) ----
    const char* gk = gKt + (size_t)key0 * (Dh * 2);
    async_copy_b128<0>(ldsKt, gk);
    async_copy_b128<2048>(ldsKt, gk);
    async_copy_b128<4096>(ldsKt, gk);
    async_copy_b128<6144>(ldsKt, gk);
    // ---- V block: b128 loads (1 base + imm offsets), frag-order scatter ----
    const uint4* pV = pV0 + (size_t)key0 * Dh / 8;
    #pragma unroll
    for (int it = 0; it < 4; ++it) {
      DW4 w; w.u4 = pV[it * 128];            // +it*16 key rows
      int k = vk0 + it * 16;
      int c = k >> 5, kk = k & 31;
      int laneBase = 16 * (kk >> 4), j = kk & 15;
      #pragma unroll
      for (int u = 0; u < 8; ++u) {
        int d = vd0 + u;
        lVf[d >> 4][c][laneBase + (d & 15)][j] = w.h[u];
      }
    }
    if (key0 + 64 < Lseq)
      __builtin_prefetch(Kh + headBase + (size_t)(key0 + 64 + (tid >> 1)) * Dh, 0, 3);
    wait_asynccnt0();          // K block landed in LDS
    __syncthreads();

    // ---- scores S[16 x 64] = Q K^T, scaled + masked ----
    const int* pmk = pm + key0;
    float sc[4][8];
    #pragma unroll
    for (int nt = 0; nt < 4; ++nt) {
      v8f s = {};
      // B[d][n] = K[key0+nt*16+n][d]: per-lane contiguous 32B (2 x b128)
      v16h bK0 = *(const v16h*)&lK[nt * 16 + lm][lh * 16];
      v16h bK1 = *(const v16h*)&lK[nt * 16 + lm][32 + lh * 16];
      s = __builtin_amdgcn_wmma_f32_16x16x32_f16(
          false, aQ[0], false, bK0, (short)0, s, false, false);
      s = __builtin_amdgcn_wmma_f32_16x16x32_f16(
          false, aQ[1], false, bK1, (short)0, s, false, false);
      #pragma unroll
      for (int v = 0; v < 8; ++v) {
        float val = s[v] * 0.125f;  // 1/sqrt(D), D=64
        if (pmk[v * Lseq + nt * 16] == 0) val = -1.0e9f;   // imm offsets
        sc[nt][v] = val;
      }
    }

    // ---- online softmax (rows live in 16-lane halves at fixed v) ----
    float pr[4][8];
    #pragma unroll
    for (int v = 0; v < 8; ++v) {
      float rmax = fmaxf(fmaxf(sc[0][v], sc[1][v]), fmaxf(sc[2][v], sc[3][v]));
      #pragma unroll
      for (int off = 1; off < 16; off <<= 1)
        rmax = fmaxf(rmax, __shfl_xor(rmax, off, 32));
      float mnew = fmaxf(mrow[v], rmax);
      float corr = __expf(mrow[v] - mnew);
      float rsum = 0.0f;
      #pragma unroll
      for (int nt = 0; nt < 4; ++nt) {
        pr[nt][v] = __expf(sc[nt][v] - mnew);
        rsum += pr[nt][v];
      }
      #pragma unroll
      for (int off = 1; off < 16; off <<= 1)
        rsum += __shfl_xor(rsum, off, 32);
      lrow[v] = lrow[v] * corr + rsum;
      mrow[v] = mnew;
      #pragma unroll
      for (int t = 0; t < 4; ++t) O[t][v] = O[t][v] * corr;
    }

    // ---- P: C-layout regs -> per-wave LDS -> A-layout fragments ----
    #pragma unroll
    for (int nt = 0; nt < 4; ++nt)
      #pragma unroll
      for (int v = 0; v < 8; ++v)
        lP[wv][v + 8 * lh][nt * 16 + lm] = (_Float16)pr[nt][v];
    // same-wave LDS ops are in-order; compiler inserts s_wait_dscnt

    #pragma unroll
    for (int c = 0; c < 2; ++c) {
      v8h plo = *(const v8h*)&lP[wv][lm][c * 32 + lh * 16];
      v8h phi = *(const v8h*)&lP[wv][lm][c * 32 + lh * 16 + 8];
      v16h aP = concat8(plo, phi);
      v16h bV[4];
      #pragma unroll
      for (int t = 0; t < 4; ++t)            // 8 x ds_load_b128, one wait
        bV[t] = *(const v16h*)&lVf[t][c][ln][0];
      #pragma unroll
      for (int t = 0; t < 4; ++t)
        O[t] = __builtin_amdgcn_wmma_f32_16x16x32_f16(
            false, aP, false, bV[t], (short)0, O[t], false, false);
    }
    __syncthreads();  // protect lK/lVf before next block's staging
  }

  // epilogue: normalize, merge heads into Z[b, l, h*D + d] (f16)
  #pragma unroll
  for (int v = 0; v < 8; ++v) {
    float inv = 1.0f / lrow[v];
    int qr = q0 + v + 8 * lh;
    #pragma unroll
    for (int t = 0; t < 4; ++t) {
      int d = h * Dh + t * 16 + lm;
      Z[((size_t)b * Lseq + qr) * Cdim + d] = (_Float16)(O[t][v] * inv);
    }
  }
}

// ---------------------------------------------------------------------------
// Kernel 3: out = Z @ WO + bO, Z is f16 [M,C], out f32 [M,C].
// X tile staged via global_load_async_to_lds_b128 (equal 16B strides).
// ---------------------------------------------------------------------------
__global__ void __launch_bounds__(128)
out_proj_kernel(const _Float16* __restrict__ X, const float* __restrict__ W,
                const float* __restrict__ bias, float* __restrict__ out)
{
  __shared__ _Float16 lX[64][64];
  __shared__ _Float16 lWf[4][2][32][16];
  const int tid = threadIdx.x;
  const int wv = tid >> 5, ln = tid & 31, lh = ln >> 4, lm = ln & 15;
  const int m0 = blockIdx.x * 64;
  const int n0 = blockIdx.y * 64;
  // thread covers 64 contiguous LDS bytes of row (tid>>1), half (tid&1)
  const unsigned ldsXt = (unsigned)(size_t)&lX[0][0] + (unsigned)tid * 64u;
  const int xr = tid >> 1, xc = (tid & 1) * 32;      // row, f16 col base
  const int sr = tid >> 4, sc4 = (tid & 15) << 2;

  v8f acc[4] = {};

  for (int k0 = 0; k0 < Cdim; k0 += 64) {
    // stage X tile 64x64 f16 via async LDS loads: 1 base + imm offsets
    const char* gx = (const char*)&X[(size_t)(m0 + xr) * Cdim + k0 + xc];
    async_copy_b128<0>(ldsXt, gx);
    async_copy_b128<16>(ldsXt, gx);
    async_copy_b128<32>(ldsXt, gx);
    async_copy_b128<48>(ldsXt, gx);
    // stage W tile (f32 -> f16) scattered into B-fragment order
    const float4* pW = (const float4*)&W[(size_t)(k0 + sr) * Cdim + n0 + sc4];
    #pragma unroll
    for (int it = 0; it < 8; ++it) {
      float4 f = pW[it * (8 * Cdim / 4)];
      int r = sr + it * 8;
      int c = r >> 5, kk = r & 31;
      int laneBase = 16 * (kk >> 4), j = kk & 15;
      float fv[4] = { f.x, f.y, f.z, f.w };
      #pragma unroll
      for (int u = 0; u < 4; ++u) {
        int n = sc4 + u;
        lWf[n >> 4][c][laneBase + (n & 15)][j] = (_Float16)fv[u];
      }
    }
    if (k0 + 64 < Cdim)
      __builtin_prefetch(&W[(size_t)(k0 + 64 + (tid >> 1)) * Cdim + n0], 0, 3);
    wait_asynccnt0();
    __syncthreads();

    #pragma unroll
    for (int c = 0; c < 2; ++c) {
      v8h alo = *(const v8h*)&lX[wv * 16 + lm][c * 32 + lh * 16];
      v8h ahi = *(const v8h*)&lX[wv * 16 + lm][c * 32 + lh * 16 + 8];
      v16h a = concat8(alo, ahi);
      v16h bfr[4];
      #pragma unroll
      for (int t = 0; t < 4; ++t)
        bfr[t] = *(const v16h*)&lWf[t][c][ln][0];
      #pragma unroll
      for (int t = 0; t < 4; ++t)
        acc[t] = __builtin_amdgcn_wmma_f32_16x16x32_f16(
            false, a, false, bfr[t], (short)0, acc[t], false, false);
    }
    __syncthreads();
  }

  #pragma unroll
  for (int t = 0; t < 4; ++t) {
    #pragma unroll
    for (int v = 0; v < 8; ++v) {
      int m = m0 + wv * 16 + v + 8 * lh;
      int n = n0 + t * 16 + lm;
      out[(size_t)m * Cdim + n] = acc[t][v] + bias[n];
    }
  }
}

// ---------------------------------------------------------------------------
extern "C" void kernel_launch(void* const* d_in, const int* in_sizes, int n_in,
                              void* d_out, int out_size, void* d_ws, size_t ws_size,
                              hipStream_t stream)
{
  const float* query = (const float*)d_in[0];
  const float* key   = (const float*)d_in[1];
  const float* value = (const float*)d_in[2];
  const int*   mask  = (const int*)  d_in[3];
  const float* WQ = (const float*)d_in[4];
  const float* bQ = (const float*)d_in[5];
  const float* WK = (const float*)d_in[6];
  const float* bK = (const float*)d_in[7];
  const float* WV = (const float*)d_in[8];
  const float* bV = (const float*)d_in[9];
  const float* WO = (const float*)d_in[10];
  const float* bO = (const float*)d_in[11];
  float* out = (float*)d_out;

  const size_t headElems = (size_t)Bsz * Hn * Lseq * Dh;  // 4M f16 = 8 MB
  _Float16* Qb = (_Float16*)d_ws;
  _Float16* Kb = Qb + headElems;
  _Float16* Vb = Kb + headElems;
  _Float16* Zb = Vb + headElems;   // total 32 MB workspace

  dim3 blk(128);
  dim3 gGemm(Mtot / 64, Cdim / 64);          // 64 x 16 blocks
  qkv_proj_kernel<<<gGemm, blk, 0, stream>>>(query, WQ, bQ, Qb);
  qkv_proj_kernel<<<gGemm, blk, 0, stream>>>(key,   WK, bK, Kb);
  qkv_proj_kernel<<<gGemm, blk, 0, stream>>>(value, WV, bV, Vb);

  dim3 gAttn(Bsz * Hn, Lseq / 64);           // 32 x 32 blocks
  attn_kernel<<<gAttn, blk, 0, stream>>>(Qb, Kb, Vb, mask, Zb);

  out_proj_kernel<<<gGemm, blk, 0, stream>>>(Zb, WO, bO, out);
}
